// WindowAttention3D_52252572123369
// MI455X (gfx1250) — matrix operand order repaired and checked by reading, not compiled, not run on hardware
//
#include <hip/hip_runtime.h>

typedef __attribute__((ext_vector_type(16))) _Float16 v16h;
typedef __attribute__((ext_vector_type(8)))  float    v8f;
typedef __attribute__((ext_vector_type(4)))  unsigned u32x4;
typedef __attribute__((ext_vector_type(8)))  int      i32x8;
typedef __attribute__((ext_vector_type(4)))  int      i32x4;

#define B_WIN   64
#define NTOK    392
#define NPAD    416          // NTOK padded to multiple of 32
#define NHEADS  16
#define DIM     512
#define HD      32
#define NW      8
#define SCALE_Q 0.17677669529663687f   // 32^-0.5

// ---- WMMA operand loaders (layouts per CDNA5 ISA 7.12.2, f16, wave32) ----
static __device__ __forceinline__ v16h loadA16(const _Float16* p, int hi) {
  union { v16h v; uint4 u[2]; } t;
  t.u[0] = *(const uint4*)(p + hi * 8);
  t.u[1] = *(const uint4*)(p + 16 + hi * 8);
  return t.v;
}
static __device__ __forceinline__ v16h loadB16(const _Float16* p, int hi) {
  union { v16h v; uint4 u[2]; } t;
  t.u[0] = *(const uint4*)(p + hi * 8);
  t.u[1] = *(const uint4*)(p + 16 + hi * 8);
  return t.v;
}
static __device__ __forceinline__ v8f wmma_f16(v16h a, v16h b, v8f c) {
  return __builtin_amdgcn_wmma_f32_16x16x32_f16(false, a, false, b, (short)0, c,
                                                false, false);
}

// ---- CDNA5 async copy: global -> LDS, 16B per lane, tracked by ASYNCcnt ----
static __device__ __forceinline__ void async_b128(unsigned lds_addr,
                                                  unsigned voff,
                                                  unsigned long long sbase) {
  asm volatile("global_load_async_to_lds_b128 %0, %1, %2"
               :: "v"(lds_addr), "v"(voff), "s"(sbase)
               : "memory");
}
static __device__ __forceinline__ void wait_async0() {
  asm volatile("s_wait_asynccnt 0x0" ::: "memory");
}
static __device__ __forceinline__ unsigned lds_off32(const void* p) {
  // generic LDS address: [31:0] is the wave-relative LDS byte offset
  return (unsigned)(unsigned long long)p;
}

// ---- CDNA5 Tensor Data Mover: 2D tile (global -> LDS), TENSORcnt ----
// D# per ISA 8.3/8.4: group0 = {count=1, lds_addr, global_addr, type=2},
// group1 = {data_size=2B, tensor dims (huge: tiles always in-bounds),
//           tile_dim0/1, dim0 stride}.  Wave-uniform; EXEC ignored.
static __device__ __forceinline__ void tdm_load_2d(unsigned lds_addr,
                                                   const void* gaddr,
                                                   unsigned tile_d0_elems,
                                                   unsigned tile_d1_rows,
                                                   unsigned stride_elems) {
  unsigned long long ga = (unsigned long long)gaddr;
  u32x4 g0;
  g0[0] = 1u;                                   // count=1 (valid), user mode
  g0[1] = lds_addr;                              // lds_addr [63:32]
  g0[2] = (unsigned)ga;                          // global_addr [95:64]
  g0[3] = (unsigned)(ga >> 32) | 0x80000000u;    // addr[56:32] | type=2
  const unsigned TD = 0x7FFFFFFFu;               // tensor_dim0/1: no clipping
  i32x8 g1;
  g1[0] = 0x00010000;                            // data_size=1 (2 bytes)
  g1[1] = (int)((TD & 0xFFFFu) << 16);           // tensor_dim0 lo
  g1[2] = (int)((TD >> 16) | ((TD & 0xFFFFu) << 16)); // td0 hi | td1 lo
  g1[3] = (int)((TD >> 16) | (tile_d0_elems << 16));  // td1 hi | tile_dim0
  g1[4] = (int)(tile_d1_rows & 0xFFFFu);         // tile_dim1 (tile_dim2=0)
  g1[5] = (int)stride_elems;                     // tensor_dim0_stride lo
  g1[6] = 0;                                     // stride hi / dim1_stride lo
  g1[7] = 0;
  i32x4 z4 = {0, 0, 0, 0};
#if defined(__clang_major__) && (__clang_major__ >= 23)
  i32x8 z8 = {0, 0, 0, 0, 0, 0, 0, 0};
  __builtin_amdgcn_tensor_load_to_lds(g0, g1, z4, z4, z8, 0);
#else
  __builtin_amdgcn_tensor_load_to_lds(g0, g1, z4, z4, 0);
#endif
}

// ---- elementwise f32 -> f16 ----
__global__ void k_f32_to_f16(const float* __restrict__ in,
                             _Float16* __restrict__ out, int n) {
  int i = blockIdx.x * blockDim.x + threadIdx.x;
  int stride = gridDim.x * blockDim.x;
  for (; i < n; i += stride) out[i] = (_Float16)in[i];
}

// ---- shared 128x128 LDS-tiled GEMM core (K=DIM, k-step 32, 8 waves) ----
// A panel (128x32) filled by TDM (wave 0); B panel (128x32) filled by
// per-lane async-to-LDS loads.  Double buffered, one barrier per k-step.
static __device__ __forceinline__ void fill_A_tdm(const _Float16* A, int row0,
                                                  int k0, _Float16* Abuf) {
  tdm_load_2d(lds_off32(Abuf), (const void*)(A + (size_t)row0 * DIM + k0),
              /*tile_d0=*/32, /*tile_d1=*/128, /*stride=*/DIM);
}
static __device__ __forceinline__ void fill_B_async(const _Float16* B, int col0,
                                                    int k0, int tid,
                                                    _Float16* Bbuf) {
  unsigned ldsB = lds_off32(Bbuf);
  unsigned long long bb = (unsigned long long)B;
  #pragma unroll
  for (int i = 0; i < 2; ++i) {
    int chunk = tid + i * 256;          // 0..511, 16B each = 8KB panel
    int r = chunk >> 2, p = chunk & 3;
    unsigned la = (unsigned)(r * 64 + p * 16);
    unsigned vb = (unsigned)((col0 + r) * DIM + k0) * 2u + (unsigned)(p * 16);
    async_b128(ldsB + la, vb, bb);
  }
}

static __device__ __forceinline__ void gemm_core_128x128(
    const _Float16* __restrict__ A, const _Float16* __restrict__ B,
    int row0, int col0, int tid,
    _Float16 (*Ab)[128 * 32], _Float16 (*Bb)[128 * 32],
    v8f acc[2][4]) {
  const int KT = DIM / 32;              // 16 k-steps
  int lane = tid & 31;
  int wid  = tid >> 5;
  int waveM = wid >> 1, waveN = wid & 1;
  int col = lane & 15, hi = lane >> 4;

  if (wid == 0) fill_A_tdm(A, row0, 0, Ab[0]);
  fill_B_async(B, col0, 0, tid, Bb[0]);

  for (int kt = 0; kt < KT; ++kt) {
    int buf = kt & 1;
    wait_async0();
    if (wid == 0) __builtin_amdgcn_s_wait_tensorcnt(0);
    __syncthreads();                    // panel[buf] published; prev reads done
    if (kt + 1 < KT) {
      if (wid == 0) fill_A_tdm(A, row0, (kt + 1) * 32, Ab[1 - buf]);
      fill_B_async(B, col0, (kt + 1) * 32, tid, Bb[1 - buf]);
    }

    const _Float16* As = Ab[buf];
    const _Float16* Bs = Bb[buf];
    v16h a0 = loadA16(As + (waveM * 32 + col) * 32, hi);
    v16h a1 = loadA16(As + (waveM * 32 + 16 + col) * 32, hi);
    #pragma unroll
    for (int tj = 0; tj < 4; ++tj) {
      v16h b = loadB16(Bs + (waveN * 64 + tj * 16 + col) * 32, hi);
      acc[0][tj] = wmma_f16(a0, b, acc[0][tj]);
      acc[1][tj] = wmma_f16(a1, b, acc[1][tj]);
    }
  }
}

// ---- QKV GEMM: [25088,512] x [1536,512]^T -> scatter q/k/v layouts ----
__global__ void __launch_bounds__(256)
k_qkv_gemm(const _Float16* __restrict__ xh,
           const _Float16* __restrict__ wh,
           _Float16* __restrict__ qh,   // [b][h][NPAD][HD]
           _Float16* __restrict__ kh,   // [b][h][NPAD][HD]
           _Float16* __restrict__ vt) { // [b][h][HD][NPAD]
  __shared__ _Float16 Ab[2][128 * 32];
  __shared__ _Float16 Bb[2][128 * 32];
  const int NB = (3 * DIM) / 128;       // 12
  int wg = blockIdx.x;
  int row0 = (wg / NB) * 128;
  int col0 = (wg % NB) * 128;
  int tid = threadIdx.x;

  v8f acc[2][4] = {};
  gemm_core_128x128(xh, wh, row0, col0, tid, Ab, Bb, acc);

  int lane = tid & 31, wid = tid >> 5;
  int waveM = wid >> 1, waveN = wid & 1;
  int col = lane & 15, hi = lane >> 4;
  #pragma unroll
  for (int ti = 0; ti < 2; ++ti) {
    #pragma unroll
    for (int tj = 0; tj < 4; ++tj) {
      int gcol  = col0 + waveN * 64 + tj * 16 + col;
      int which = gcol >> 9;            // 0=q 1=k 2=v
      int c     = gcol & 511;
      int hh    = c >> 5, d = c & 31;
      #pragma unroll
      for (int e = 0; e < 8; ++e) {
        int grow = row0 + waveM * 32 + ti * 16 + hi * 8 + e;
        int bb = grow / NTOK, nn = grow % NTOK;
        float v = acc[ti][tj][e];
        if (which == 0) {
          qh[(((size_t)bb * NHEADS + hh) * NPAD + nn) * HD + d] =
              (_Float16)(v * SCALE_Q);
        } else if (which == 1) {
          kh[(((size_t)bb * NHEADS + hh) * NPAD + nn) * HD + d] = (_Float16)v;
        } else {
          vt[(((size_t)bb * NHEADS + hh) * HD + d) * NPAD + nn] = (_Float16)v;
        }
      }
    }
  }
}

// ---- output projection GEMM: [25088,512] x [512,512]^T + bias -> fp32 ----
__global__ void __launch_bounds__(256)
k_proj_gemm(const _Float16* __restrict__ oh,
            const _Float16* __restrict__ wh,
            const float* __restrict__ bias,
            float* __restrict__ out) {
  __shared__ _Float16 Ab[2][128 * 32];
  __shared__ _Float16 Bb[2][128 * 32];
  const int NB = DIM / 128;             // 4
  int wg = blockIdx.x;
  int row0 = (wg / NB) * 128;
  int col0 = (wg % NB) * 128;
  int tid = threadIdx.x;

  v8f acc[2][4] = {};
  gemm_core_128x128(oh, wh, row0, col0, tid, Ab, Bb, acc);

  int lane = tid & 31, wid = tid >> 5;
  int waveM = wid >> 1, waveN = wid & 1;
  int col = lane & 15, hi = lane >> 4;
  #pragma unroll
  for (int ti = 0; ti < 2; ++ti) {
    #pragma unroll
    for (int tj = 0; tj < 4; ++tj) {
      int gcol = col0 + waveN * 64 + tj * 16 + col;
      float bv = bias[gcol];
      #pragma unroll
      for (int e = 0; e < 8; ++e) {
        int grow = row0 + waveM * 32 + ti * 16 + hi * 8 + e;
        out[(size_t)grow * DIM + gcol] = acc[ti][tj][e] + bv;
      }
    }
  }
}

// ---- fused attention: one wave per (b, h, 16-row query block) ----
__global__ void k_attn(const _Float16* __restrict__ qh,
                       const _Float16* __restrict__ kh,
                       const _Float16* __restrict__ vt,
                       const float* __restrict__ mask,   // [NW][N][N]
                       const float* __restrict__ table,  // [2535][16]
                       const int*  __restrict__ relidx,  // [N][N]
                       _Float16* __restrict__ oh) {      // [b*N][DIM]
  __shared__ float    S[16][432];
  __shared__ _Float16 P[16][424];

  int wg = blockIdx.x;
  int rb = wg % 25;
  int h  = (wg / 25) % NHEADS;
  int b  = wg / (25 * NHEADS);
  int n0 = rb * 16;
  int lane = threadIdx.x & 31;
  int col = lane & 15, hi = lane >> 4;

  const _Float16* qbase = qh + ((size_t)(b * NHEADS + h) * NPAD) * HD;
  const _Float16* kbase = kh + ((size_t)(b * NHEADS + h) * NPAD) * HD;
  const _Float16* vbase = vt + ((size_t)(b * NHEADS + h) * HD) * NPAD;
  const float*    mrow  = mask + (size_t)(b % NW) * NTOK * NTOK;

  // Q block as A-matrix (rows n0..n0+15, K = head_dim = 32)
  v16h QA = loadA16(qbase + (size_t)(n0 + col) * HD, hi);

  // S = (q*scale) K^T + bias + mask  -> LDS fp32
  for (int mt = 0; mt < NPAD / 16; ++mt) {
    int m0 = mt * 16;
    v16h KB = loadB16(kbase + (size_t)(m0 + col) * HD, hi);
    v8f s = {};
    s = wmma_f16(QA, KB, s);
    int m = m0 + col;
    #pragma unroll
    for (int e = 0; e < 8; ++e) {
      int r = hi * 8 + e;
      int n = n0 + r;
      float val = -1e30f;
      if (m < NTOK && n < NTOK) {
        int idx = relidx[n * NTOK + m];
        val = s[e] + table[idx * NHEADS + h] + mrow[n * NTOK + m];
      }
      S[r][m] = val;
    }
  }
  __syncthreads();

  // softmax: 2 lanes per row (lane, lane+16) handle 208 columns each
  {
    int r  = col;
    int mb = hi * 208;
    float mx = -3.0e30f;
    for (int m = mb; m < mb + 208; ++m) mx = fmaxf(mx, S[r][m]);
    mx = fmaxf(mx, __shfl_xor(mx, 16, 32));
    float sum = 0.f;
    for (int m = mb; m < mb + 208; ++m) {
      float ev = __expf(S[r][m] - mx);
      S[r][m] = ev;
      sum += ev;
    }
    sum += __shfl_xor(sum, 16, 32);
    float inv = 1.f / sum;
    for (int m = mb; m < mb + 208; ++m)
      P[r][m] = (_Float16)(S[r][m] * inv);
  }
  __syncthreads();

  // O = P V   (K = NPAD = 13 steps of 32, N = HD = two 16-col tiles)
  v8f o0 = {}, o1 = {};
  for (int kt = 0; kt < NPAD / 32; ++kt) {
    int kk = kt * 32;
    v16h PA = loadA16(&P[col][kk], hi);
    v16h V0 = loadB16(vbase + (size_t)(0 + col)  * NPAD + kk, hi);
    v16h V1 = loadB16(vbase + (size_t)(16 + col) * NPAD + kk, hi);
    o0 = wmma_f16(PA, V0, o0);
    o1 = wmma_f16(PA, V1, o1);
  }
  #pragma unroll
  for (int e = 0; e < 8; ++e) {
    int n = n0 + hi * 8 + e;
    if (n < NTOK) {
      size_t base = ((size_t)b * NTOK + n) * DIM + (size_t)h * HD;
      oh[base + col]      = (_Float16)o0[e];
      oh[base + 16 + col] = (_Float16)o1[e];
    }
  }
}

extern "C" void kernel_launch(void* const* d_in, const int* in_sizes, int n_in,
                              void* d_out, int out_size, void* d_ws,
                              size_t ws_size, hipStream_t stream) {
  const float* x      = (const float*)d_in[0];
  const float* mask   = (const float*)d_in[1];
  const float* qkv_w  = (const float*)d_in[2];
  const float* proj_w = (const float*)d_in[3];
  const float* proj_b = (const float*)d_in[4];
  const float* table  = (const float*)d_in[5];
  const int*   relidx = (const int*)d_in[6];

  const size_t M = (size_t)B_WIN * NTOK;   // 25088 = 196 * 128

  char* ws = (char*)d_ws;
  size_t off = 0;
  auto take = [&](size_t bytes) -> void* {
    void* p = ws + off;
    off += (bytes + 255) & ~(size_t)255;
    return p;
  };

  _Float16* xh      = (_Float16*)take(M * DIM * 2);
  _Float16* qkv_wh  = (_Float16*)take((size_t)3 * DIM * DIM * 2);
  _Float16* proj_wh = (_Float16*)take((size_t)DIM * DIM * 2);
  const size_t qkv_bytes = (size_t)B_WIN * NHEADS * NPAD * HD * 2;
  _Float16* qh = (_Float16*)take(3 * qkv_bytes);   // q,k,vt contiguous
  _Float16* kh = qh + qkv_bytes / 2;
  _Float16* vt = kh + qkv_bytes / 2;
  _Float16* oh = (_Float16*)take(M * DIM * 2);

  // zero padded q/k/v so padding is deterministic
  hipMemsetAsync(qh, 0, 3 * qkv_bytes, stream);

  k_f32_to_f16<<<2048, 256, 0, stream>>>(x, xh, (int)(M * DIM));
  k_f32_to_f16<<<512, 256, 0, stream>>>(qkv_w, qkv_wh, 3 * DIM * DIM);
  k_f32_to_f16<<<256, 256, 0, stream>>>(proj_w, proj_wh, DIM * DIM);

  {
    int wgs = (int)(M / 128) * ((3 * DIM) / 128);  // 196*12 = 2352
    k_qkv_gemm<<<wgs, 256, 0, stream>>>(xh, qkv_wh, qh, kh, vt);
  }

  k_attn<<<B_WIN * NHEADS * 25, 32, 0, stream>>>(qh, kh, vt, mask, table,
                                                 relidx, oh);

  {
    int wgs = (int)(M / 128) * (DIM / 128);        // 196*4 = 784
    k_proj_gemm<<<wgs, 256, 0, stream>>>(oh, proj_wh, proj_b, (float*)d_out);
  }
}
